// ModelODIN_56684978373081
// MI455X (gfx1250) — matrix operand order repaired and checked by reading, not compiled, not run on hardware
//
#include <hip/hip_runtime.h>

// ---------------- problem constants ----------------
#define BB   1024      // batch
#define CCLS 100       // classes
#define CPAD 112       // classes padded to multiple of 16
#define DD   512       // feature dim
#define NIN  3072      // input dim (3*32*32)
#define EPS  0.0014f

typedef __attribute__((ext_vector_type(16))) __bf16 v16bf;
typedef __attribute__((ext_vector_type(8)))  float  v8f;

struct alignas(16) U16x8 { unsigned int w[4]; };
union AFrag { v16bf v; U16x8 h[2]; };

enum { EPI_F32 = 0, EPI_RELU = 1, EPI_XADV = 2 };

// CDNA5 async global->LDS copy: no VGPR staging, tracked by ASYNCcnt.
// (VDST = 32-bit LDS byte address VGPR, VADDR = 64-bit global address.)
__device__ __forceinline__ void async_copy_b128(const __bf16* g, unsigned lds) {
    asm volatile("global_load_async_to_lds_b128 %0, %1, off"
                 :: "v"(lds), "v"(g)
                 : "memory");
}
__device__ __forceinline__ void wait_async0() {
    asm volatile("s_wait_asynccnt 0x0" ::: "memory");
}

// ---------------- bf16 WMMA GEMM ----------------
// C[MxN] = A[MxK] * B'[NxK]   (B is COLUMN-major: B'[n*ldb + k])
// block = 256 threads = 8 waves; tile 64x128, K-step 32.
// Double-buffered LDS fed by GLOBAL_LOAD_ASYNC_TO_LDS_B128 (no staging VGPRs,
// no spills); one s_wait_asynccnt + one barrier per K-step; next tile's async
// copies overlap this tile's WMMAs. Each wave: 2x2 v_wmma_f32_16x16x32_bf16.
template <int EPI, bool GUARD>
__global__ void __launch_bounds__(256)
gemm_bf16_kernel(const __bf16* __restrict__ A, const __bf16* __restrict__ Bc,
                 const float* __restrict__ bias, const float* __restrict__ xsrc,
                 void* __restrict__ Cout,
                 int M, int N, int K, int lda, int ldb, int ldc)
{
    constexpr int BM = 64, BN = 128, BK = 32, LDSW = BK + 8;   // +8 bf16 pad
    constexpr int WAVES_M = 2, WN = 32, NI = 2;                // 8 waves: 2(M) x 4(N)
    constexpr int AITER = (BM * 4) / 256;                      // 16B chunks / thread
    constexpr int BITER = (BN * 4) / 256;

    __shared__ __bf16 As[2][BM][LDSW];
    __shared__ __bf16 Bs[2][BN][LDSW];

    const int tid  = threadIdx.x;
    const int lane = tid & 31;
    const int wave = tid >> 5;
    const int wm   = wave % WAVES_M;
    const int wn   = wave / WAVES_M;
    const int bRow = blockIdx.y * BM;
    const int bCol = blockIdx.x * BN;

    v8f zero = {};
    v8f acc[2][NI];
#pragma unroll
    for (int i = 0; i < 2; ++i)
#pragma unroll
        for (int j = 0; j < NI; ++j) acc[i][j] = zero;

    // per-thread copy mapping: one 16B chunk per (operand, i); K-major for both
    const __bf16* aG[AITER]; unsigned aL[AITER][2]; bool aOk[AITER];
#pragma unroll
    for (int i = 0; i < AITER; ++i) {
        int linear = tid + i * 256;
        int row = linear >> 2, ch = (linear & 3) * 8;
        aOk[i] = !GUARD || (bRow + row) < M;
        aG[i]  = A + (size_t)(bRow + row) * lda + ch;
        aL[i][0] = (unsigned)(size_t)&As[0][row][ch];
        aL[i][1] = (unsigned)(size_t)&As[1][row][ch];
    }
    const __bf16* bG[BITER]; unsigned bL[BITER][2]; bool bOk[BITER];
#pragma unroll
    for (int i = 0; i < BITER; ++i) {
        int linear = tid + i * 256;
        int row = linear >> 2, ch = (linear & 3) * 8;
        bOk[i] = !GUARD || (bCol + row) < N;
        bG[i]  = Bc + (size_t)(bCol + row) * ldb + ch;
        bL[i][0] = (unsigned)(size_t)&Bs[0][row][ch];
        bL[i][1] = (unsigned)(size_t)&Bs[1][row][ch];
    }

    if constexpr (GUARD) {   // pad rows never get copied into -> keep them zero
        __bf16* pa = &As[0][0][0];
        for (int i = tid; i < 2 * BM * LDSW; i += 256) pa[i] = (__bf16)0.f;
        __bf16* pb = &Bs[0][0][0];
        for (int i = tid; i < 2 * BN * LDSW; i += 256) pb[i] = (__bf16)0.f;
        __syncthreads();
    }

    auto issue = [&](int kk, int buf) {
#pragma unroll
        for (int i = 0; i < AITER; ++i)
            if (aOk[i]) async_copy_b128(aG[i] + kk, aL[i][buf]);
#pragma unroll
        for (int i = 0; i < BITER; ++i)
            if (bOk[i]) async_copy_b128(bG[i] + kk, bL[i][buf]);
    };

    // ISA 7.12.2 fragment lane mapping (wave32, 16-bit A 16x32 / B 32x16)
    const int am  = lane & 15;            // row (A) / col (B) within 16-tile
    const int kbA = (lane >> 4) * 8;      // A: elems 0-7 -> K kbA..,  8-15 -> K kbA+16..
    const int kbB = (lane >> 4) * 8;      // B: elems 0-7 -> K kbB..,  8-15 -> K kbB+16..

    auto mma = [&](int buf) {
        AFrag af[2], bfr[NI];
#pragma unroll
        for (int mi = 0; mi < 2; ++mi) {
            const __bf16* ar = &As[buf][wm * 32 + mi * 16 + am][0];
            af[mi].h[0] = *(const U16x8*)(ar + kbA);
            af[mi].h[1] = *(const U16x8*)(ar + kbA + 16);
        }
#pragma unroll
        for (int ni = 0; ni < NI; ++ni) {
            const __bf16* br = &Bs[buf][wn * WN + ni * 16 + am][0];
            bfr[ni].h[0] = *(const U16x8*)(br + kbB);
            bfr[ni].h[1] = *(const U16x8*)(br + kbB + 16);
        }
#pragma unroll
        for (int mi = 0; mi < 2; ++mi)
#pragma unroll
            for (int ni = 0; ni < NI; ++ni)
                acc[mi][ni] = __builtin_amdgcn_wmma_f32_16x16x32_bf16(
                    false, af[mi].v, false, bfr[ni].v,
                    (short)0, acc[mi][ni], false, false);
    };

    const int nTiles = K / BK;            // K is always a multiple of 32 here
    issue(0, 0);
    int t = 0;
    for (; t < nTiles - 1; ++t) {         // steady state: branch-free body
        wait_async0();                    // this wave's tile-t copies landed
        __syncthreads();                  // everyone's landed; buf^1 readers done
        issue((t + 1) * BK, (t + 1) & 1); // overlap next tile with compute
        mma(t & 1);
    }
    wait_async0();                        // peeled last tile
    __syncthreads();
    mma(t & 1);

    // ---- epilogue: C/D layout (v8f elem r -> row r + 8*(lane>>4), col lane&15) ----
    const int cm = 8 * (lane >> 4);
    const int cn = lane & 15;
#pragma unroll
    for (int mi = 0; mi < 2; ++mi) {
#pragma unroll
        for (int ni = 0; ni < NI; ++ni) {
            int gcol = bCol + wn * WN + ni * 16 + cn;
#pragma unroll
            for (int r = 0; r < 8; ++r) {
                int grow = bRow + wm * 32 + mi * 16 + cm + r;
                if (!GUARD || (grow < M && gcol < N)) {
                    float a = acc[mi][ni][r];
                    size_t idx = (size_t)grow * ldc + gcol;
                    if constexpr (EPI == EPI_F32) {
                        ((float*)Cout)[idx] = a;
                    } else if constexpr (EPI == EPI_RELU) {
                        float v = a + bias[gcol];
                        ((__bf16*)Cout)[idx] = (__bf16)(v > 0.f ? v : 0.f);
                    } else { // EPI_XADV: x_adv = clip(x + eps*grad, 0, 1), stored bf16
                        float v = xsrc[idx] + EPS * a;
                        v = v < 0.f ? 0.f : (v > 1.f ? 1.f : v);
                        ((__bf16*)Cout)[idx] = (__bf16)v;
                    }
                }
            }
        }
    }
}

// ---------------- helper kernels ----------------
__global__ void cvt_bf16_kernel(const float* __restrict__ src, __bf16* __restrict__ dst, int n) {
    int i = blockIdx.x * blockDim.x + threadIdx.x;
    if (i < n) dst[i] = (__bf16)src[i];
}

// dst[Cc x R] = bf16( src[R x Cc] ^T )   (32x32 LDS tile transpose)
__global__ void transpose_cvt_kernel(const float* __restrict__ src, __bf16* __restrict__ dst,
                                     int R, int Cc) {
    __shared__ float tile[32][33];
    int bx = blockIdx.x * 32, by = blockIdx.y * 32;
    int tx = threadIdx.x, ty = threadIdx.y;
#pragma unroll
    for (int i = 0; i < 32; i += 8) {
        int r = by + ty + i, c = bx + tx;
        tile[ty + i][tx] = (r < R && c < Cc) ? src[(size_t)r * Cc + c] : 0.f;
    }
    __syncthreads();
#pragma unroll
    for (int i = 0; i < 32; i += 8) {
        int r = bx + ty + i, c = by + tx;
        if (r < Cc && c < R) dst[(size_t)r * R + c] = (__bf16)tile[tx][ty + i];
    }
}

// SM[c][d] = sum_k mu[c][k]*S[k][d]  (f32) + bf16 row-major copy (= col-major SM^T)
__global__ void sm_kernel(const float* __restrict__ mu, const float* __restrict__ S,
                          float* __restrict__ SMf, __bf16* __restrict__ SMbf) {
    int d = blockIdx.x * blockDim.x + threadIdx.x;
    int c = blockIdx.y;
    if (d >= DD) return;
    float acc = 0.f;
    if (c < CCLS) {
        for (int k = 0; k < DD; ++k) acc = fmaf(mu[(size_t)c * DD + k], S[(size_t)k * DD + d], acc);
    }
    SMf[(size_t)c * DD + d] = acc;
    SMbf[(size_t)c * DD + d] = (__bf16)acc;
}

// const_c = mu_c . SM_c
__global__ void constc_kernel(const float* __restrict__ mu, const float* __restrict__ SMf,
                              float* __restrict__ constc) {
    int c = blockIdx.x;
    __shared__ float red[256];
    float acc = 0.f;
    if (c < CCLS)
        for (int d = threadIdx.x; d < DD; d += 256)
            acc = fmaf(mu[(size_t)c * DD + d], SMf[(size_t)c * DD + d], acc);
    red[threadIdx.x] = acc;
    __syncthreads();
    for (int s = 128; s > 0; s >>= 1) {
        if (threadIdx.x < s) red[threadIdx.x] += red[threadIdx.x + s];
        __syncthreads();
    }
    if (threadIdx.x == 0) constc[c] = red[0];
}

// fSf[b] = feat[b] . G[b]
__global__ void fsf_kernel(const __bf16* __restrict__ feat, const float* __restrict__ G,
                           float* __restrict__ fSf) {
    int b = blockIdx.x;
    float acc = 0.f;
    for (int d = threadIdx.x; d < DD; d += 256)
        acc = fmaf((float)feat[(size_t)b * DD + d], G[(size_t)b * DD + d], acc);
    __shared__ float red[256];
    red[threadIdx.x] = acc;
    __syncthreads();
    for (int s = 128; s > 0; s >>= 1) {
        if (threadIdx.x < s) red[threadIdx.x] += red[threadIdx.x + s];
        __syncthreads();
    }
    if (threadIdx.x == 0) fSf[b] = red[0];
}

// c*[b] = argmax_c (2*cross[b][c] - const_c)     (fSf drops out of the argmax)
__global__ void argmax_kernel(const float* __restrict__ cross, const float* __restrict__ constc,
                              int* __restrict__ cstar) {
    int b = blockIdx.x, tid = threadIdx.x;
    float best = -3.4e38f; int bi = 0;
    for (int c = tid; c < CCLS; c += 128) {
        float s = 2.f * cross[(size_t)b * CPAD + c] - constc[c];
        if (s > best) { best = s; bi = c; }
    }
    __shared__ float bv[128]; __shared__ int bx[128];
    bv[tid] = best; bx[tid] = bi;
    __syncthreads();
    for (int s = 64; s > 0; s >>= 1) {
        if (tid < s && bv[tid + s] > bv[tid]) { bv[tid] = bv[tid + s]; bx[tid] = bx[tid + s]; }
        __syncthreads();
    }
    if (tid == 0) cstar[b] = bx[0];
}

// dL/dz = -2*(G - SM[c*]) * 1[z>0]   (relu mask via feat>0), stored bf16
__global__ void dldz_kernel(const __bf16* __restrict__ feat, const float* __restrict__ G,
                            const float* __restrict__ SMf, const int* __restrict__ cstar,
                            __bf16* __restrict__ dLdz) {
    int i = blockIdx.x * blockDim.x + threadIdx.x;
    if (i >= BB * DD) return;
    int b = i >> 9, d = i & (DD - 1);
    float v = 0.f;
    if ((float)feat[i] > 0.f)
        v = -2.f * (G[i] - SMf[(size_t)cstar[b] * DD + d]);
    dLdz[i] = (__bf16)v;
}

// out[b][c] = -(fSf[b] - 2*cross[b][c] + const_c)
__global__ void scores_kernel(const float* __restrict__ fSf, const float* __restrict__ cross,
                              const float* __restrict__ constc, float* __restrict__ out) {
    int i = blockIdx.x * blockDim.x + threadIdx.x;
    if (i >= BB * CCLS) return;
    int b = i / CCLS, c = i - b * CCLS;
    out[i] = -(fSf[b] - 2.f * cross[(size_t)b * CPAD + c] + constc[c]);
}

// ---------------- host orchestration ----------------
extern "C" void kernel_launch(void* const* d_in, const int* in_sizes, int n_in,
                              void* d_out, int out_size, void* d_ws, size_t ws_size,
                              hipStream_t stream) {
    const float* x   = (const float*)d_in[0];   // [1024,3,32,32]
    const float* W   = (const float*)d_in[1];   // [3072,512]
    const float* bv  = (const float*)d_in[2];   // [512]
    const float* mu  = (const float*)d_in[3];   // [100,512]
    const float* S   = (const float*)d_in[4];   // [512,512]
    float* out = (float*)d_out;                 // [1024,100]
    (void)in_sizes; (void)n_in; (void)out_size; (void)ws_size;

    char* ws = (char*)d_ws;
    size_t off = 0;
    auto take = [&](size_t nbytes) -> void* {
        void* p = (void*)(ws + off);
        off = (off + nbytes + 255) & ~(size_t)255;
        return p;
    };
    __bf16* Xbf    = (__bf16*)take((size_t)BB * NIN * 2);
    __bf16* Wbf    = (__bf16*)take((size_t)NIN * DD * 2);   // = col-major B for backward GEMM
    __bf16* WbfT   = (__bf16*)take((size_t)DD * NIN * 2);   // = col-major B for forward GEMM
    __bf16* Sbf    = (__bf16*)take((size_t)DD * DD * 2);    // S symmetric: row-major == col-major
    float*  SMf    = (float*) take((size_t)CPAD * DD * 4);
    __bf16* SMbf   = (__bf16*)take((size_t)CPAD * DD * 2);  // = col-major SM^T for cross GEMM
    float*  constc = (float*) take((size_t)CPAD * 4);
    __bf16* featbf = (__bf16*)take((size_t)BB * DD * 2);
    float*  Gf     = (float*) take((size_t)BB * DD * 4);
    float*  fSf    = (float*) take((size_t)BB * 4);
    float*  crossf = (float*) take((size_t)BB * CPAD * 4);
    int*    cstar  = (int*)   take((size_t)BB * 4);
    __bf16* dLdzbf = (__bf16*)take((size_t)BB * DD * 2);
    __bf16* xadvbf = (__bf16*)take((size_t)BB * NIN * 2);

    // --- precompute (bf16 operands, Mahalanobis expansion terms) ---
    cvt_bf16_kernel<<<(BB * NIN + 255) / 256, 256, 0, stream>>>(x, Xbf, BB * NIN);
    cvt_bf16_kernel<<<(NIN * DD + 255) / 256, 256, 0, stream>>>(W, Wbf, NIN * DD);
    transpose_cvt_kernel<<<dim3(DD / 32, NIN / 32), dim3(32, 8), 0, stream>>>(W, WbfT, NIN, DD);
    cvt_bf16_kernel<<<(DD * DD + 255) / 256, 256, 0, stream>>>(S, Sbf, DD * DD);
    sm_kernel<<<dim3(DD / 256, CPAD), 256, 0, stream>>>(mu, S, SMf, SMbf);
    constc_kernel<<<CPAD, 256, 0, stream>>>(mu, SMf, constc);

    // --- pass 1: feat = relu(X W + b); G = feat S; cross = feat SM^T; argmax ---
    gemm_bf16_kernel<EPI_RELU, false><<<dim3(DD / 128, BB / 64), 256, 0, stream>>>(
        Xbf, WbfT, bv, nullptr, featbf, BB, DD, NIN, NIN, NIN, DD);
    gemm_bf16_kernel<EPI_F32, false><<<dim3(DD / 128, BB / 64), 256, 0, stream>>>(
        featbf, Sbf, nullptr, nullptr, Gf, BB, DD, DD, DD, DD, DD);
    gemm_bf16_kernel<EPI_F32, true><<<dim3(1, BB / 64), 256, 0, stream>>>(
        featbf, SMbf, nullptr, nullptr, crossf, BB, CPAD, DD, DD, DD, CPAD);
    argmax_kernel<<<BB, 128, 0, stream>>>(crossf, constc, cstar);

    // --- analytic FGSM gradient + fused x_adv = clip(x + eps*g) ---
    dldz_kernel<<<(BB * DD + 255) / 256, 256, 0, stream>>>(featbf, Gf, SMf, cstar, dLdzbf);
    gemm_bf16_kernel<EPI_XADV, false><<<dim3(NIN / 128, BB / 64), 256, 0, stream>>>(
        dLdzbf, Wbf, nullptr, x, xadvbf, BB, NIN, DD, DD, DD, NIN);

    // --- pass 2 on x_adv ---
    gemm_bf16_kernel<EPI_RELU, false><<<dim3(DD / 128, BB / 64), 256, 0, stream>>>(
        xadvbf, WbfT, bv, nullptr, featbf, BB, DD, NIN, NIN, NIN, DD);
    gemm_bf16_kernel<EPI_F32, false><<<dim3(DD / 128, BB / 64), 256, 0, stream>>>(
        featbf, Sbf, nullptr, nullptr, Gf, BB, DD, DD, DD, DD, DD);
    fsf_kernel<<<BB, 256, 0, stream>>>(featbf, Gf, fSf);
    gemm_bf16_kernel<EPI_F32, true><<<dim3(1, BB / 64), 256, 0, stream>>>(
        featbf, SMbf, nullptr, nullptr, crossf, BB, CPAD, DD, DD, DD, CPAD);
    scores_kernel<<<(BB * CCLS + 255) / 256, 256, 0, stream>>>(fSf, crossf, constc, out);
}
